// GraphBasedController_68719477424
// MI455X (gfx1250) — matrix operand order repaired and checked by reading, not compiled, not run on hardware
//
#include <hip/hip_runtime.h>
#include <hip/hip_bf16.h>

// ---------- types ----------
typedef __attribute__((ext_vector_type(16))) __bf16        v16bf;
typedef __attribute__((ext_vector_type(16))) unsigned short v16u;
typedef __attribute__((ext_vector_type(8)))  float          v8f;
typedef __attribute__((ext_vector_type(4)))  unsigned int   v4u;

union Frag16 { v16u u; v16bf b; v4u q[2]; };

static __device__ __forceinline__ unsigned short f2bf(float f) {
    unsigned int x = __float_as_uint(f);
    return (unsigned short)((x + 0x7fffu + ((x >> 16) & 1u)) >> 16);
}

#define BB 8
#define RR 128
#define HH 256
#define IN_DIM 512
#define LL 3

// ---------- sigmoid adjacency + row sums ----------
__global__ void k_adj(const float* __restrict__ adj, float* __restrict__ A,
                      float* __restrict__ rowsum) {
    __shared__ float red[RR];
    int i = blockIdx.x, j = threadIdx.x;
    float v = 1.0f / (1.0f + __expf(-adj[i * RR + j]));
    if (i == j) v = 0.0f;
    A[i * RR + j] = v;
    red[j] = v;
    __syncthreads();
    for (int s = 64; s > 0; s >>= 1) { if (j < s) red[j] += red[j + s]; __syncthreads(); }
    if (j == 0) rowsum[i] = red[0];
}

// ---------- fp32 [K,N] -> bf16 transposed [N,K] ----------
__global__ void k_convT(const float* __restrict__ W, unsigned short* __restrict__ WT,
                        int K, int N) {
    int idx = blockIdx.x * blockDim.x + threadIdx.x;
    if (idx >= K * N) return;
    int n = idx / K, k = idx % K;
    WT[idx] = f2bf(W[(size_t)k * N + n]);
}

// ---------- encoder: h0 = x@We + be, broadcast over R ----------
__global__ void k_encode(const float* __restrict__ x, const float* __restrict__ We,
                         const float* __restrict__ be, float* __restrict__ h,
                         unsigned short* __restrict__ hbf, unsigned short* __restrict__ cat) {
    int b = blockIdx.x, c = threadIdx.x;
    float s = be[c];
    for (int k = 0; k < IN_DIM; ++k) s = fmaf(x[b * IN_DIM + k], We[(size_t)k * HH + c], s);
    unsigned short sb = f2bf(s);
    for (int r = 0; r < RR; ++r) {
        size_t m = (size_t)b * RR + r;
        h[m * HH + c]   = s;
        hbf[m * HH + c] = sb;
        cat[m * 2 * HH + c] = sb;
    }
}

// ---------- WMMA bf16 GEMM: C[M,N] = A[M,K] @ BT[N,K]^T (+bias, +rowscale*bias2, relu) ----------
__global__ __launch_bounds__(128)
void k_gemm_bf16(const unsigned short* __restrict__ A, int lda,
                 const unsigned short* __restrict__ BT,
                 const float* __restrict__ bias,
                 const float* __restrict__ rowscale, const float* __restrict__ bias2, int rs_mod,
                 float* __restrict__ Cf, int ldcf,
                 unsigned short* __restrict__ Cb, int ldcb, int cb_off,
                 int M, int N, int K, int relu) {
    const int lane = threadIdx.x & 31;
    const int wave = threadIdx.x >> 5;
    const int half = lane >> 4;
    const int l16  = lane & 15;
    const int m0 = blockIdx.y * 64 + wave * 16;
    const int n0 = blockIdx.x * 64;
    if (m0 >= M) return;

    v8f z = {0.f, 0.f, 0.f, 0.f, 0.f, 0.f, 0.f, 0.f};
    v8f acc[4] = {z, z, z, z};

    const unsigned short* arow = A + (size_t)(m0 + l16) * lda;
    for (int k0 = 0; k0 < K; k0 += 32) {
        Frag16 af;
        const unsigned short* pa = arow + k0 + 8 * half;     // K = k0+8h .. +7
        af.q[0] = *(const v4u*)pa;                            // then K = k0+16+8h .. +7
        af.q[1] = *(const v4u*)(pa + 16);
#pragma unroll
        for (int t = 0; t < 4; ++t) {
            Frag16 bf;
            const unsigned short* pb = BT + (size_t)(n0 + t * 16 + l16) * K + k0 + 8 * half;
            bf.q[0] = *(const v4u*)pb;
            bf.q[1] = *(const v4u*)(pb + 16);
            acc[t] = __builtin_amdgcn_wmma_f32_16x16x32_bf16(
                         false, af.b, false, bf.b, (short)0, acc[t], false, false);
        }
    }

#pragma unroll
    for (int t = 0; t < 4; ++t) {
        int n = n0 + t * 16 + l16;
        float bcol  = bias ? bias[n] : 0.0f;
        float b2col = rowscale ? bias2[n] : 0.0f;
#pragma unroll
        for (int r = 0; r < 8; ++r) {
            int m = m0 + half * 8 + r;
            float v = acc[t][r] + bcol;
            if (rowscale) v = fmaf(rowscale[m % rs_mod], b2col, v);
            if (relu) v = fmaxf(v, 0.0f);
            if (Cf) Cf[(size_t)m * ldcf + n] = v;
            if (Cb) Cb[(size_t)m * ldcb + cb_off + n] = f2bf(v);
        }
    }
}

// ---------- fused message + weighted aggregation over senders j ----------
// aggpre[b,i,c] = sum_j A[i,j] * relu(a[b,i,c] + bb[b,j,c] + b1[c])
// bb chunks are staged global->LDS with CDNA5 async-to-LDS DMA (ASYNCcnt).
__global__ __launch_bounds__(256)
void k_msg_agg(const float* __restrict__ a, const float* __restrict__ bb,
               const float* __restrict__ b1, const float* __restrict__ A,
               float* __restrict__ aggpre, unsigned short* __restrict__ aggpre_bf) {
    __shared__ __align__(16) float sBB[32 * HH];   // 32 KB
    __shared__ float sA[16 * RR];                  //  8 KB
    int b  = blockIdx.x >> 3;
    int i0 = (blockIdx.x & 7) * 16;
    int c  = threadIdx.x;

    for (int t = threadIdx.x; t < 16 * RR; t += 256)
        sA[t] = A[(size_t)(i0 + (t >> 7)) * RR + (t & 127)];

    float ai[16], acc[16];
#pragma unroll
    for (int il = 0; il < 16; ++il) {
        ai[il]  = a[((size_t)b * RR + i0 + il) * HH + c];
        acc[il] = 0.0f;
    }
    float b1c = b1[c];

    for (int j0 = 0; j0 < RR; j0 += 32) {
        __syncthreads();    // everyone done reading previous sBB chunk
        {
            const float* gsrc = bb + ((size_t)b * RR + j0) * HH;
            // 32*256 floats = 32KB chunk; 16B per async transfer per lane
#pragma unroll
            for (int t4 = 0; t4 < 32 * HH; t4 += 256 * 4) {
                int e = t4 + threadIdx.x * 4;
                unsigned ldsa = (unsigned)(size_t)(&sBB[e]);
                const float* gp = gsrc + e;
                asm volatile("global_load_async_to_lds_b128 %0, %1, off"
                             :: "v"(ldsa), "v"(gp)
                             : "memory");
            }
            asm volatile("s_wait_asynccnt 0" ::: "memory");
        }
        __syncthreads();
        for (int jj = 0; jj < 32; ++jj) {
            float bv = sBB[jj * HH + c] + b1c;
            int j = j0 + jj;
#pragma unroll
            for (int il = 0; il < 16; ++il) {
                float pre = fmaxf(ai[il] + bv, 0.0f);
                acc[il] = fmaf(sA[il * RR + j], pre, acc[il]);
            }
        }
    }
#pragma unroll
    for (int il = 0; il < 16; ++il) {
        size_t m = (size_t)b * RR + i0 + il;
        aggpre[m * HH + c]    = acc[il];
        aggpre_bf[m * HH + c] = f2bf(acc[il]);
    }
}

// ---------- residual + layernorm; refresh f32/bf16/cat copies of h ----------
__global__ void k_res_ln(float* __restrict__ h, const float* __restrict__ u,
                         const float* __restrict__ g, const float* __restrict__ bta,
                         unsigned short* __restrict__ hbf, unsigned short* __restrict__ cat) {
    __shared__ float red[HH];
    int m = blockIdx.x, c = threadIdx.x;
    float v = h[(size_t)m * HH + c] + u[(size_t)m * HH + c];
    red[c] = v; __syncthreads();
    for (int s = 128; s > 0; s >>= 1) { if (c < s) red[c] += red[c + s]; __syncthreads(); }
    float mu = red[0] * (1.0f / HH);
    __syncthreads();
    float d = v - mu;
    red[c] = d * d; __syncthreads();
    for (int s = 128; s > 0; s >>= 1) { if (c < s) red[c] += red[c + s]; __syncthreads(); }
    float var = red[0] * (1.0f / HH);
    float o = d * rsqrtf(var + 1e-5f) * g[c] + bta[c];
    h[(size_t)m * HH + c]   = o;
    unsigned short ob = f2bf(o);
    hbf[(size_t)m * HH + c] = ob;
    cat[(size_t)m * 2 * HH + c] = ob;
}

// ---------- multihead attention, flash-style online softmax ----------
__global__ __launch_bounds__(128)
void k_attn(const float* __restrict__ q, const float* __restrict__ k,
            const float* __restrict__ v, unsigned short* __restrict__ obf) {
    __shared__ float sK[32 * 64];
    __shared__ float sV[32 * 64];
    int b  = blockIdx.x >> 2;
    int hd = blockIdx.x & 3;
    int r  = threadIdx.x;                 // query row 0..127
    size_t base = (size_t)b * RR * HH + hd * 64;

    float qr[64];
#pragma unroll
    for (int d = 0; d < 64; ++d) qr[d] = q[base + (size_t)r * HH + d];
    float mmax = -1e30f, ssum = 0.0f;
    float acc[64];
#pragma unroll
    for (int d = 0; d < 64; ++d) acc[d] = 0.0f;
    const float scale = 0.125f;           // 1/sqrt(64)

    for (int j0 = 0; j0 < RR; j0 += 32) {
        __syncthreads();
        for (int t = threadIdx.x; t < 32 * 64; t += 128) {
            int jj = t >> 6, d = t & 63;
            sK[t] = k[base + (size_t)(j0 + jj) * HH + d];
            sV[t] = v[base + (size_t)(j0 + jj) * HH + d];
        }
        __syncthreads();
        float sc[32];
#pragma unroll
        for (int jj = 0; jj < 32; ++jj) {
            float s = 0.0f;
            for (int d = 0; d < 64; ++d) s = fmaf(qr[d], sK[jj * 64 + d], s);
            sc[jj] = s * scale;
        }
        float nm = mmax;
#pragma unroll
        for (int jj = 0; jj < 32; ++jj) nm = fmaxf(nm, sc[jj]);
        float corr = __expf(mmax - nm);
        ssum *= corr;
#pragma unroll
        for (int d = 0; d < 64; ++d) acc[d] *= corr;
        for (int jj = 0; jj < 32; ++jj) {
            float p = __expf(sc[jj] - nm);
            ssum += p;
            for (int d = 0; d < 64; ++d) acc[d] = fmaf(p, sV[jj * 64 + d], acc[d]);
        }
        mmax = nm;
    }
    float inv = 1.0f / ssum;
    size_t ob = ((size_t)b * RR + r) * HH + hd * 64;
#pragma unroll
    for (int d = 0; d < 64; ++d) obf[ob + d] = f2bf(acc[d] * inv);
}

// ---------- mean over nodes ----------
__global__ void k_mean(const float* __restrict__ pooled, float* __restrict__ g) {
    int b = blockIdx.x, c = threadIdx.x;
    float s = 0.0f;
    for (int r = 0; r < RR; ++r) s += pooled[((size_t)b * RR + r) * HH + c];
    g[b * HH + c] = s * (1.0f / RR);
}

// ---------- readout MLP + softmax (M=8: scalar path) ----------
__global__ void k_readout(const float* __restrict__ g, const float* __restrict__ W1,
                          const float* __restrict__ b1, const float* __restrict__ W2,
                          const float* __restrict__ b2, float* __restrict__ out) {
    __shared__ float sg[HH];
    __shared__ float sh[HH];
    __shared__ float red[RR];
    int b = blockIdx.x, t = threadIdx.x;
    sg[t] = g[b * HH + t];
    __syncthreads();
    float s = b1[t];
    for (int k = 0; k < HH; ++k) s = fmaf(sg[k], W1[(size_t)k * HH + t], s);
    sh[t] = fmaxf(s, 0.0f);
    __syncthreads();
    float logit = 0.0f;
    if (t < RR) {
        logit = b2[t];
        for (int k = 0; k < HH; ++k) logit = fmaf(sh[k], W2[(size_t)k * RR + t], logit);
        red[t] = logit;
    }
    __syncthreads();
    for (int s2 = 64; s2 > 0; s2 >>= 1) { if (t < s2) red[t] = fmaxf(red[t], red[t + s2]); __syncthreads(); }
    float mx = red[0];
    __syncthreads();
    float e = 0.0f;
    if (t < RR) { e = __expf(logit - mx); red[t] = e; }
    __syncthreads();
    for (int s2 = 64; s2 > 0; s2 >>= 1) { if (t < s2) red[t] += red[t + s2]; __syncthreads(); }
    if (t < RR) out[b * RR + t] = e / red[0];
}

// ================= host orchestration =================
extern "C" void kernel_launch(void* const* d_in, const int* in_sizes, int n_in,
                              void* d_out, int out_size, void* d_ws, size_t ws_size,
                              hipStream_t stream) {
    const float* x      = (const float*)d_in[0];
    const float* We     = (const float*)d_in[1];
    const float* be     = (const float*)d_in[2];
    const float* msg_W1 = (const float*)d_in[3];
    const float* msg_b1 = (const float*)d_in[4];
    const float* msg_W2 = (const float*)d_in[5];
    const float* msg_b2 = (const float*)d_in[6];
    const float* upd_W1 = (const float*)d_in[7];
    const float* upd_b1 = (const float*)d_in[8];
    const float* upd_W2 = (const float*)d_in[9];
    const float* upd_b2 = (const float*)d_in[10];
    const float* ln_g   = (const float*)d_in[11];
    const float* ln_b   = (const float*)d_in[12];
    const float* adj    = (const float*)d_in[13];
    const float* Wq = (const float*)d_in[14]; const float* bq = (const float*)d_in[15];
    const float* Wk = (const float*)d_in[16]; const float* bk = (const float*)d_in[17];
    const float* Wv = (const float*)d_in[18]; const float* bv = (const float*)d_in[19];
    const float* Wo = (const float*)d_in[20]; const float* bo = (const float*)d_in[21];
    const float* ro_W1 = (const float*)d_in[22]; const float* ro_b1 = (const float*)d_in[23];
    const float* ro_W2 = (const float*)d_in[24]; const float* ro_b2 = (const float*)d_in[25];

    const int M = BB * RR;          // 1024
    size_t off = 0;
    auto alloc = [&](size_t bytes) -> char* {
        char* p = (char*)d_ws + off;
        off = (off + bytes + 255) & ~(size_t)255;
        return p;
    };
    float* h_f32   = (float*)alloc((size_t)M * HH * 4);
    float* u_f32   = (float*)alloc((size_t)M * HH * 4);   // also pooled
    float* a_f32   = (float*)alloc((size_t)M * HH * 4);   // also q
    float* bb_f32  = (float*)alloc((size_t)M * HH * 4);   // also k
    float* ag_f32  = (float*)alloc((size_t)M * HH * 4);   // also v
    float* g_f32   = (float*)alloc((size_t)BB * HH * 4);
    float* Aw      = (float*)alloc((size_t)RR * RR * 4);
    float* rowsum  = (float*)alloc((size_t)RR * 4);
    unsigned short* h_bf    = (unsigned short*)alloc((size_t)M * HH * 2);
    unsigned short* cat_bf  = (unsigned short*)alloc((size_t)M * 2 * HH * 2);
    unsigned short* ag_bf   = (unsigned short*)alloc((size_t)M * HH * 2);
    unsigned short* u1_bf   = (unsigned short*)alloc((size_t)M * HH * 2);
    unsigned short* at_bf   = (unsigned short*)alloc((size_t)M * HH * 2);
    unsigned short* msgW1aT = (unsigned short*)alloc((size_t)LL * HH * HH * 2);
    unsigned short* msgW1bT = (unsigned short*)alloc((size_t)LL * HH * HH * 2);
    unsigned short* msgW2T  = (unsigned short*)alloc((size_t)LL * HH * HH * 2);
    unsigned short* updW1T  = (unsigned short*)alloc((size_t)LL * 2 * HH * HH * 2);
    unsigned short* updW2T  = (unsigned short*)alloc((size_t)LL * HH * HH * 2);
    unsigned short* WqT = (unsigned short*)alloc((size_t)HH * HH * 2);
    unsigned short* WkT = (unsigned short*)alloc((size_t)HH * HH * 2);
    unsigned short* WvT = (unsigned short*)alloc((size_t)HH * HH * 2);
    unsigned short* WoT = (unsigned short*)alloc((size_t)HH * HH * 2);
    if (off > ws_size) return;   // workspace too small; bail safely

    auto convT = [&](const float* W, unsigned short* WT, int K, int N) {
        int tot = K * N;
        k_convT<<<(tot + 255) / 256, 256, 0, stream>>>(W, WT, K, N);
    };
    auto gemm = [&](const unsigned short* A, int lda, const unsigned short* BT,
                    const float* bias, const float* rsc, const float* b2, int rs_mod,
                    float* Cf, unsigned short* Cb, int ldcb, int cb_off,
                    int Mm, int N, int K, int relu) {
        dim3 grd(N / 64, Mm / 64);
        k_gemm_bf16<<<grd, 128, 0, stream>>>(A, lda, BT, bias, rsc, b2, rs_mod,
                                             Cf, N, Cb, ldcb, cb_off, Mm, N, K, relu);
    };

    // ---- one-time prep ----
    k_adj<<<RR, RR, 0, stream>>>(adj, Aw, rowsum);
    for (int l = 0; l < LL; ++l) {
        const float* W1l = msg_W1 + (size_t)l * 2 * HH * HH;
        convT(W1l,                   msgW1aT + (size_t)l * HH * HH, HH, HH);   // rows 0..H-1
        convT(W1l + (size_t)HH * HH, msgW1bT + (size_t)l * HH * HH, HH, HH);   // rows H..2H-1
        convT(msg_W2 + (size_t)l * HH * HH, msgW2T + (size_t)l * HH * HH, HH, HH);
        convT(upd_W1 + (size_t)l * 2 * HH * HH, updW1T + (size_t)l * 2 * HH * HH, 2 * HH, HH);
        convT(upd_W2 + (size_t)l * HH * HH, updW2T + (size_t)l * HH * HH, HH, HH);
    }
    convT(Wq, WqT, HH, HH); convT(Wk, WkT, HH, HH);
    convT(Wv, WvT, HH, HH); convT(Wo, WoT, HH, HH);

    // ---- encoder ----
    k_encode<<<BB, HH, 0, stream>>>(x, We, be, h_f32, h_bf, cat_bf);

    // ---- graph layers ----
    for (int l = 0; l < LL; ++l) {
        const unsigned short* W1aT = msgW1aT + (size_t)l * HH * HH;
        const unsigned short* W1bT = msgW1bT + (size_t)l * HH * HH;
        const unsigned short* W2T  = msgW2T  + (size_t)l * HH * HH;
        const unsigned short* U1T  = updW1T  + (size_t)l * 2 * HH * HH;
        const unsigned short* U2T  = updW2T  + (size_t)l * HH * HH;
        // a = h@W1a ; bb = h@W1b    (bias b1 folded into message kernel)
        gemm(h_bf, HH, W1aT, nullptr, nullptr, nullptr, 1, a_f32, nullptr, 0, 0, M, HH, HH, 0);
        gemm(h_bf, HH, W1bT, nullptr, nullptr, nullptr, 1, bb_f32, nullptr, 0, 0, M, HH, HH, 0);
        // aggpre = sum_j A[i,j]*relu(a_i + bb_j + b1)
        k_msg_agg<<<BB * (RR / 16), 256, 0, stream>>>(a_f32, bb_f32, msg_b1 + (size_t)l * HH,
                                                      Aw, ag_f32, ag_bf);
        // agg = aggpre@W2 + rowsum[i]*b2  -> bf16 straight into cat[:, H:2H]
        gemm(ag_bf, HH, W2T, nullptr, rowsum, msg_b2 + (size_t)l * HH, RR,
             nullptr, cat_bf, 2 * HH, HH, M, HH, HH, 0);
        // u1 = relu(cat(h,agg)@Wu1 + b1)  (K=512)
        gemm(cat_bf, 2 * HH, U1T, upd_b1 + (size_t)l * HH, nullptr, nullptr, 1,
             nullptr, u1_bf, HH, 0, M, HH, 2 * HH, 1);
        // u = u1@Wu2 + b2
        gemm(u1_bf, HH, U2T, upd_b2 + (size_t)l * HH, nullptr, nullptr, 1,
             u_f32, nullptr, 0, 0, M, HH, HH, 0);
        // h = LN(h + u)
        k_res_ln<<<M, HH, 0, stream>>>(h_f32, u_f32, ln_g + (size_t)l * HH,
                                       ln_b + (size_t)l * HH, h_bf, cat_bf);
    }

    // ---- attention ----
    gemm(h_bf, HH, WqT, bq, nullptr, nullptr, 1, a_f32,  nullptr, 0, 0, M, HH, HH, 0);
    gemm(h_bf, HH, WkT, bk, nullptr, nullptr, 1, bb_f32, nullptr, 0, 0, M, HH, HH, 0);
    gemm(h_bf, HH, WvT, bv, nullptr, nullptr, 1, ag_f32, nullptr, 0, 0, M, HH, HH, 0);
    k_attn<<<BB * 4, 128, 0, stream>>>(a_f32, bb_f32, ag_f32, at_bf);
    gemm(at_bf, HH, WoT, bo, nullptr, nullptr, 1, u_f32, nullptr, 0, 0, M, HH, HH, 0);

    // ---- pool + readout ----
    k_mean<<<BB, HH, 0, stream>>>(u_f32, g_f32);
    k_readout<<<BB, HH, 0, stream>>>(g_f32, ro_W1, ro_b1, ro_W2, ro_b2, (float*)d_out);
}
